// GATBlock_3058016714958
// MI455X (gfx1250) — matrix operand order repaired and checked by reading, not compiled, not run on hardware
//
#include <hip/hip_runtime.h>

typedef unsigned short u16;
typedef __attribute__((ext_vector_type(4)))  unsigned int v4u;
typedef __attribute__((ext_vector_type(8)))  unsigned int v8u;
typedef __attribute__((ext_vector_type(16))) __bf16      v16bf;
typedef __attribute__((ext_vector_type(8)))  float       v8f;

#define B_DIM 2048
#define N_DIM 17
#define D_DIM 512
#define H_DIM 8
#define HID_DIM 2048
#define M_ROWS (B_DIM * N_DIM)   // 34816, multiple of 64

__device__ __forceinline__ u16 f2bf(float f) {
    unsigned int u = __float_as_uint(f);
    u += 0x7FFFu + ((u >> 16) & 1u);   // round-to-nearest-even
    return (u16)(u >> 16);
}
__device__ __forceinline__ float bf2f(u16 h) {
    return __uint_as_float(((unsigned int)h) << 16);
}
__device__ __forceinline__ v16bf pack16(v4u lo, v4u hi) {
    v8u u;
    u[0] = lo.x; u[1] = lo.y; u[2] = lo.z; u[3] = lo.w;
    u[4] = hi.x; u[5] = hi.y; u[6] = hi.z; u[7] = hi.w;
    return __builtin_bit_cast(v16bf, u);
}
__device__ __forceinline__ float gelu_tanh(float x) {
    const float c = 0.7978845608028654f;
    float t = tanhf(c * (x + 0.044715f * x * x * x));
    return 0.5f * x * (1.0f + t);
}
// async copy 16B/lane from global to LDS (ASYNCcnt-tracked, no VGPR round-trip)
__device__ __forceinline__ void async_g2l_b128(unsigned lds_off, const void* gptr) {
    asm volatile("global_load_async_to_lds_b128 %0, %1, off"
                 :: "v"(lds_off), "v"((unsigned long long)(uintptr_t)gptr)
                 : "memory");
}
__device__ __forceinline__ void wait_async0() {
    asm volatile("s_wait_asynccnt 0" ::: "memory");
}

// ---------------------------------------------------------------------------
// Weight transpose + fp32->bf16:  Wt[n*K + k] = bf16(W[k*N + n])
// ---------------------------------------------------------------------------
__global__ __launch_bounds__(256) void transpose_to_bf16(
    const float* __restrict__ W, u16* __restrict__ Wt, int K, int N) {
    size_t idx = (size_t)blockIdx.x * 256 + threadIdx.x;
    size_t total = (size_t)K * N;
    if (idx >= total) return;
    int n = (int)(idx / K);
    int k = (int)(idx % K);
    Wt[idx] = f2bf(W[(size_t)k * N + n]);
}

// ---------------------------------------------------------------------------
// LayerNorm over D=512, one wave32 per row (16 elems/lane), output bf16
// ---------------------------------------------------------------------------
__global__ __launch_bounds__(256) void ln_bf16_kernel(
    const float* __restrict__ x, const float* __restrict__ g,
    const float* __restrict__ b, u16* __restrict__ out) {
    const int wave = threadIdx.x >> 5;
    const int lane = threadIdx.x & 31;
    const size_t row = (size_t)blockIdx.x * 8 + wave;
    const float* xr = x + row * D_DIM;
    float vals[16];
    float s = 0.f, s2 = 0.f;
#pragma unroll
    for (int i = 0; i < 16; ++i) {
        float v = xr[lane + (i << 5)];
        vals[i] = v; s += v; s2 += v * v;
    }
#pragma unroll
    for (int off = 16; off > 0; off >>= 1) {
        s  += __shfl_xor(s,  off, 32);
        s2 += __shfl_xor(s2, off, 32);
    }
    const float mean = s * (1.f / 512.f);
    const float var  = s2 * (1.f / 512.f) - mean * mean;
    const float rstd = rsqrtf(var + 1e-5f);
    u16* orow = out + row * D_DIM;
#pragma unroll
    for (int i = 0; i < 16; ++i) {
        int c = lane + (i << 5);
        orow[c] = f2bf((vals[i] - mean) * rstd * g[c] + b[c]);
    }
}

// ---------------------------------------------------------------------------
// bf16 WMMA GEMM: C[M,N] = A[M,K] @ Wt[N,K]^T   (Wt pre-transposed)
// 256 threads = 8 waves. Block tile 64(M) x 128(N); K-tile 32.
// A tile (64x32 bf16) double-buffered in LDS via global_load_async_to_lds_b128
// (ASYNCcnt). Each wave holds 4 accumulators (4 M-subtiles x its 16-col B
// fragment) -> 4 back-to-back v_wmma_f32_16x16x32_bf16 per K-step, B fragment
// amortized 4x. Epilogue fused per MODE:
//   0: plain->bf16   1: plain->f32   2: +bias +aux1*scale ->f32+bf16
//   3: +bias+aux1+aux2 ->f32         4: +bias,GELU ->bf16   5: +bias ->f32
// Requires: M%64==0, N%128==0, K%32==0
// ---------------------------------------------------------------------------
template <int MODE>
__global__ __launch_bounds__(256) void gemm_bf16_wmma(
    const u16* __restrict__ A, const u16* __restrict__ Wt,
    int M, int N, int K,
    const float* __restrict__ bias,
    const float* __restrict__ aux1,
    const float* __restrict__ scale,
    const float* __restrict__ aux2,
    float* __restrict__ outF,
    u16* __restrict__ outB) {
    constexpr bool HAS_BIAS = (MODE >= 2);
    constexpr bool HAS_AUX1 = (MODE == 2) || (MODE == 3);
    constexpr bool AUX1_SC  = (MODE == 2);
    constexpr bool HAS_AUX2 = (MODE == 3);
    constexpr bool DO_GELU  = (MODE == 4);
    constexpr bool OUT_F32  = (MODE == 1) || (MODE == 2) || (MODE == 3) || (MODE == 5);
    constexpr bool OUT_BF   = (MODE == 0) || (MODE == 2) || (MODE == 4);

    __shared__ u16 At[2][64][40];   // 64x32 tile, padded stride (bank spread)

    const int t    = threadIdx.x;
    const int wave = t >> 5;
    const int lane = t & 31;
    const int m0   = blockIdx.y * 64;
    const int n0   = blockIdx.x * 128 + wave * 16;

    // cooperative A-tile load: thread t async-copies 16B: row t>>2, col (t&3)*8
    const int lr = t >> 2;           // 0..63
    const int lc = (t & 3) << 3;     // 0,8,16,24
    const u16* __restrict__ Asrc = A + (size_t)(m0 + lr) * K + lc;
    const unsigned lds0 = (unsigned)(uintptr_t)&At[0][lr][lc];
    const unsigned lds1 = (unsigned)(uintptr_t)&At[1][lr][lc];

    // fragment mapping (ISA 7.12.2 VGPR layouts)
    const int hi   = lane >> 4;          // half-select
    const int arow = lane & 15;          // A row within 16-row subtile
    const int bcol = n0 + (lane & 15);   // B column
    const u16* __restrict__ Bsrc = Wt + (size_t)bcol * K + (hi << 4);

    v8f acc[4];
#pragma unroll
    for (int s = 0; s < 4; ++s) acc[s] = (v8f){0.f,0.f,0.f,0.f,0.f,0.f,0.f,0.f};
    const int KT = K >> 5;

    async_g2l_b128(lds0, Asrc);
    wait_async0();
    __syncthreads();

    for (int kt = 0; kt < KT; ++kt) {
        const int cur = kt & 1;
        if (kt + 1 < KT) {
            async_g2l_b128(cur ? lds0 : lds1, Asrc + (kt + 1) * 32);
            __builtin_prefetch(Bsrc + (kt + 1) * 32, 0, 1);  // global_prefetch_b8
        }
        // B fragment: lanes 0-15 K0-15, lanes 16-31 K16-31 (contiguous in Wt)
        const u16* bp = Bsrc + kt * 32;
        v4u b0 = *(const v4u*)(bp);
        v4u b1 = *(const v4u*)(bp + 8);
        v16bf bv = pack16(b0, b1);
        // A fragments: lanes 0-15 K0-15, lanes 16-31 K16-31
#pragma unroll
        for (int s = 0; s < 4; ++s) {
            const u16* ap = &At[cur][(s << 4) + arow][0];
            v4u a0 = *(const v4u*)(ap + (hi << 4));
            v4u a1 = *(const v4u*)(ap + (hi << 4) + 8);
            acc[s] = __builtin_amdgcn_wmma_f32_16x16x32_bf16(
                false, pack16(a0, a1), false, bv, (short)0, acc[s], false, false);
        }
        wait_async0();
        __syncthreads();
    }

    // C layout: VGPR r -> M = r + 8*(lane>=16), N = n0 + (lane&15)
    const int ccol = n0 + (lane & 15);
#pragma unroll
    for (int s = 0; s < 4; ++s) {
        const int crow0 = m0 + (s << 4) + (hi << 3);
#pragma unroll
        for (int r = 0; r < 8; ++r) {
            const int row = crow0 + r;
            float v = acc[s][r];
            if constexpr (HAS_BIAS) v += bias[ccol];
            if constexpr (HAS_AUX1) {
                float a1v = aux1[(size_t)row * N + ccol];
                if constexpr (AUX1_SC) v += a1v * scale[ccol];
                else                   v += a1v;
            }
            if constexpr (HAS_AUX2) v += aux2[(size_t)row * N + ccol];
            if constexpr (DO_GELU)  v = gelu_tanh(v);
            if constexpr (OUT_F32)  outF[(size_t)row * N + ccol] = v;
            if constexpr (OUT_BF)   outB[(size_t)row * N + ccol] = f2bf(v);
        }
    }
}

// ---------------------------------------------------------------------------
// Hop-biased multi-head attention; one block per (b, head). N=17, DH=64.
// qkv is bf16 [M, 1536] (q | k | v); output o written bf16 [M, 512].
// ---------------------------------------------------------------------------
__global__ __launch_bounds__(256) void attn_kernel(
    const u16* __restrict__ qkv, const float* __restrict__ hop,
    u16* __restrict__ obf) {
    const int bh = blockIdx.x;
    const int b = bh >> 3;
    const int h = bh & 7;
    __shared__ float q[17][64], k[17][64], v[17][64];
    __shared__ float a[17][18];
    const int tid = threadIdx.x;
    const u16* base = qkv + (size_t)b * N_DIM * (3 * D_DIM) + h * 64;
    for (int idx = tid; idx < 17 * 64; idx += 256) {
        int n = idx >> 6, d = idx & 63;
        const u16* p = base + (size_t)n * (3 * D_DIM) + d;
        q[n][d] = bf2f(p[0]);
        k[n][d] = bf2f(p[512]);
        v[n][d] = bf2f(p[1024]);
    }
    __syncthreads();
    const float* hb = hop + ((size_t)b * H_DIM + h) * (N_DIM * N_DIM);
    for (int idx = tid; idx < 289; idx += 256) {
        int i = idx / 17, j = idx % 17;
        float s = 0.f;
#pragma unroll
        for (int d = 0; d < 64; ++d) s += q[i][d] * k[j][d];
        a[i][j] = s * 0.125f + hb[idx];   // 64^-0.5
    }
    __syncthreads();
    if (tid < 17) {
        float mx = -3.4e38f;
        for (int j = 0; j < 17; ++j) mx = fmaxf(mx, a[tid][j]);
        float s = 0.f;
        for (int j = 0; j < 17; ++j) { float e = __expf(a[tid][j] - mx); a[tid][j] = e; s += e; }
        float inv = 1.f / s;
        for (int j = 0; j < 17; ++j) a[tid][j] *= inv;
    }
    __syncthreads();
    for (int idx = tid; idx < 17 * 64; idx += 256) {
        int i = idx >> 6, d = idx & 63;
        float s = 0.f;
#pragma unroll
        for (int j = 0; j < 17; ++j) s += a[i][j] * v[j][d];
        obf[((size_t)b * N_DIM + i) * D_DIM + h * 64 + d] = f2bf(s);
    }
}

// ---------------------------------------------------------------------------
// Joint-dimension mixers (N=17). Thread owns one (b, d) column.
// ---------------------------------------------------------------------------
__global__ __launch_bounds__(256) void mix_hA_kernel(
    const float* __restrict__ adj, const float* __restrict__ adjres,
    const u16* __restrict__ hbf, u16* __restrict__ out) {
    __shared__ float Am[17][17];
    const int tid = threadIdx.x;
    for (int i = tid; i < 289; i += 256) Am[i / 17][i % 17] = adj[i] + adjres[i];
    __syncthreads();
    const size_t tg = (size_t)blockIdx.x * 256 + tid;
    const int b = (int)(tg >> 9);
    const int d = (int)(tg & 511);
    float hv[17];
    const u16* hp = hbf + (size_t)b * N_DIM * D_DIM + d;
#pragma unroll
    for (int j = 0; j < 17; ++j) hv[j] = bf2f(hp[j * D_DIM]);
    u16* op = out + (size_t)b * N_DIM * D_DIM + d;
#pragma unroll
    for (int i = 0; i < 17; ++i) {
        float s = 0.f;
#pragma unroll
        for (int j = 0; j < 17; ++j) s += Am[i][j] * hv[j];
        op[i * D_DIM] = f2bf(s);
    }
}

__global__ __launch_bounds__(256) void mix_pos_kernel(
    const float* __restrict__ pos, const float* __restrict__ h2,
    float* __restrict__ out) {
    __shared__ float Am[17][17];
    const int tid = threadIdx.x;
    for (int i = tid; i < 289; i += 256) Am[i / 17][i % 17] = pos[i];
    __syncthreads();
    const size_t tg = (size_t)blockIdx.x * 256 + tid;
    const int b = (int)(tg >> 9);
    const int d = (int)(tg & 511);
    float hv[17];
    const float* hp = h2 + (size_t)b * N_DIM * D_DIM + d;
#pragma unroll
    for (int j = 0; j < 17; ++j) hv[j] = hp[j * D_DIM];
    float* op = out + (size_t)b * N_DIM * D_DIM + d;
#pragma unroll
    for (int i = 0; i < 17; ++i) {
        float s = 0.f;
#pragma unroll
        for (int j = 0; j < 17; ++j) s += Am[i][j] * hv[j];
        op[i * D_DIM] = s;
    }
}

__global__ __launch_bounds__(256) void final_mix_kernel(
    const float* __restrict__ wc, const float* __restrict__ bc,
    const float* __restrict__ mm, const float* __restrict__ xnew,
    float* __restrict__ out) {
    __shared__ float Am[17][17];
    __shared__ float Bb[17];
    const int tid = threadIdx.x;
    for (int i = tid; i < 289; i += 256) Am[i / 17][i % 17] = wc[i];
    if (tid < 17) Bb[tid] = bc[tid];
    __syncthreads();
    const size_t tg = (size_t)blockIdx.x * 256 + tid;
    const int b = (int)(tg >> 9);
    const int d = (int)(tg & 511);
    float mv[17];
    const size_t rowbase = (size_t)b * N_DIM * D_DIM + d;
#pragma unroll
    for (int j = 0; j < 17; ++j) mv[j] = mm[rowbase + j * D_DIM];
#pragma unroll
    for (int i = 0; i < 17; ++i) {
        float s = 0.f;
#pragma unroll
        for (int j = 0; j < 17; ++j) s += Am[i][j] * mv[j];
        out[rowbase + i * D_DIM] = xnew[rowbase + i * D_DIM] + s + Bb[i];
    }
}

// ---------------------------------------------------------------------------
extern "C" void kernel_launch(void* const* d_in, const int* in_sizes, int n_in,
                              void* d_out, int out_size, void* d_ws, size_t ws_size,
                              hipStream_t stream) {
    (void)in_sizes; (void)n_in; (void)out_size; (void)ws_size;
    const float* x        = (const float*)d_in[0];
    const float* hop      = (const float*)d_in[1];
    const float* adj      = (const float*)d_in[2];
    const float* ln1_g    = (const float*)d_in[3];
    const float* ln1_b    = (const float*)d_in[4];
    const float* w_qkv    = (const float*)d_in[5];
    const float* w_proj   = (const float*)d_in[6];
    const float* b_proj   = (const float*)d_in[7];
    const float* gcn_w    = (const float*)d_in[8];
    const float* gcn_m    = (const float*)d_in[9];
    const float* gcn_ar   = (const float*)d_in[10];
    const float* xfeat_w  = (const float*)d_in[11];
    const float* xfeat_b  = (const float*)d_in[12];
    const float* pos_mix  = (const float*)d_in[13];
    const float* ln2_g    = (const float*)d_in[14];
    const float* ln2_b    = (const float*)d_in[15];
    const float* mlp_w1   = (const float*)d_in[16];
    const float* mlp_b1   = (const float*)d_in[17];
    const float* mlp_w2   = (const float*)d_in[18];
    const float* mlp_b2   = (const float*)d_in[19];
    const float* mlp_wc   = (const float*)d_in[20];
    const float* mlp_bc   = (const float*)d_in[21];
    float* out = (float*)d_out;

    char* ws = (char*)d_ws;
    const size_t SZ_BF = (size_t)M_ROWS * D_DIM * 2;   // 35.65 MB
    const size_t SZ_F  = (size_t)M_ROWS * D_DIM * 4;   // 71.30 MB

    u16*  hbf   = (u16*)(ws);                              // h = LN1(x), bf16
    u16*  habf  = (u16*)(ws + SZ_BF);                      // (adj+res)@h, bf16
    u16*  obf   = (u16*)(ws + 2 * SZ_BF);                  // attention out, bf16
    u16*  qkvbf = (u16*)(ws + 3 * SZ_BF);                  // [M,1536] bf16
    float* t1   = (float*)(ws + 6 * SZ_BF);                // gcn pre-mod, f32
    float* h2f  = (float*)(ws + 6 * SZ_BF + SZ_F);         // h2, f32
    u16*  h2bf  = (u16*)(ws + 6 * SZ_BF + 2 * SZ_F);       // h2, bf16
    float* xnew = (float*)(ws + 7 * SZ_BF + 2 * SZ_F);     // x after branch, f32
    char*  wp   = ws + 7 * SZ_BF + 3 * SZ_F;               // bf16 weights
    u16* wqkvT  = (u16*)wp; wp += (size_t)512 * 1536 * 2;
    u16* wprojT = (u16*)wp; wp += (size_t)512 * 512 * 2;
    u16* gcnT   = (u16*)wp; wp += (size_t)512 * 512 * 2;
    u16* xfT    = (u16*)wp; wp += (size_t)512 * 512 * 2;
    u16* w1T    = (u16*)wp; wp += (size_t)512 * 2048 * 2;
    u16* w2T    = (u16*)wp; wp += (size_t)2048 * 512 * 2;
    // Aliases (lifetimes disjoint):
    u16*  ubf   = qkvbf;          // gelu acts, [M,2048] bf16 (qkv dead post-attn)
    float* mmf  = (float*)(ws + SZ_BF);  // mlp out f32 over habf+obf (both dead)
    u16*  h2ln  = hbf;            // LN2 out (hbf dead after G1 + mix_hA)
    float* pbuf = t1;             // pos_mix@h2 (t1 dead after proj GEMM)

    const dim3 blk(256);
    const dim3 gMix(4096);                      // B*512 / 256
    const dim3 gLN(M_ROWS / 8);
    const dim3 gAttn(B_DIM * H_DIM);
    const unsigned gY = M_ROWS / 64;            // 544 row-blocks

    // --- weights -> transposed bf16
    transpose_to_bf16<<<dim3((512 * 1536 + 255) / 256), blk, 0, stream>>>(w_qkv, wqkvT, 512, 1536);
    transpose_to_bf16<<<dim3((512 * 512 + 255) / 256),  blk, 0, stream>>>(w_proj, wprojT, 512, 512);
    transpose_to_bf16<<<dim3((512 * 512 + 255) / 256),  blk, 0, stream>>>(gcn_w, gcnT, 512, 512);
    transpose_to_bf16<<<dim3((512 * 512 + 255) / 256),  blk, 0, stream>>>(xfeat_w, xfT, 512, 512);
    transpose_to_bf16<<<dim3((512 * 2048 + 255) / 256), blk, 0, stream>>>(mlp_w1, w1T, 512, 2048);
    transpose_to_bf16<<<dim3((2048 * 512 + 255) / 256), blk, 0, stream>>>(mlp_w2, w2T, 2048, 512);

    // --- h = LN1(x); hA = (adj + gcn_adj_res) @ h
    ln_bf16_kernel<<<gLN, blk, 0, stream>>>(x, ln1_g, ln1_b, hbf);
    mix_hA_kernel<<<gMix, blk, 0, stream>>>(adj, gcn_ar, hbf, habf);

    // --- qkv = h @ w_qkv  (bf16 out)
    gemm_bf16_wmma<0><<<dim3(1536 / 128, gY), blk, 0, stream>>>(
        hbf, wqkvT, M_ROWS, 1536, 512, nullptr, nullptr, nullptr, nullptr, nullptr, qkvbf);

    // --- attention with hop-path bias
    attn_kernel<<<gAttn, blk, 0, stream>>>(qkvbf, hop, obf);

    // --- t1 = hA @ gcn_w  (f32 out)
    gemm_bf16_wmma<1><<<dim3(512 / 128, gY), blk, 0, stream>>>(
        habf, gcnT, M_ROWS, 512, 512, nullptr, nullptr, nullptr, nullptr, t1, nullptr);

    // --- h2 = o @ w_proj + b_proj + t1 * gcn_m  (f32 + bf16 out)
    gemm_bf16_wmma<2><<<dim3(512 / 128, gY), blk, 0, stream>>>(
        obf, wprojT, M_ROWS, 512, 512, b_proj, t1, gcn_m, nullptr, h2f, h2bf);

    // --- p = pos_mix @ h2
    mix_pos_kernel<<<gMix, blk, 0, stream>>>(pos_mix, h2f, pbuf);

    // --- x_new = x + h2 @ xfeat_w + xfeat_b + p  (f32)
    gemm_bf16_wmma<3><<<dim3(512 / 128, gY), blk, 0, stream>>>(
        h2bf, xfT, M_ROWS, 512, 512, xfeat_b, pbuf, nullptr, x, xnew, nullptr);

    // --- LN2 + MLP
    ln_bf16_kernel<<<gLN, blk, 0, stream>>>(xnew, ln2_g, ln2_b, h2ln);
    gemm_bf16_wmma<4><<<dim3(2048 / 128, gY), blk, 0, stream>>>(
        h2ln, w1T, M_ROWS, 2048, 512, mlp_b1, nullptr, nullptr, nullptr, nullptr, ubf);
    gemm_bf16_wmma<5><<<dim3(512 / 128, gY), blk, 0, stream>>>(
        ubf, w2T, M_ROWS, 512, 2048, mlp_b2, nullptr, nullptr, nullptr, mmf, nullptr);

    // --- out = x_new + mlp_wc @ m + mlp_bc[:,None]
    final_mix_kernel<<<gMix, blk, 0, stream>>>(mlp_wc, mlp_bc, mmf, xnew, out);
}
